// GQAttentionBlock_1623497638711
// MI455X (gfx1250) — matrix operand order repaired and checked by reading, not compiled, not run on hardware
//
#include <hip/hip_runtime.h>
#include <hip/hip_bf16.h>
#include <math.h>

#define PSEQ 2048
#define DEMB 768
#define HIDD 3072
#define GKV  4
#define QPG  2
#define HDIM 96
#define NQH  8      // GKV*QPG query heads
#define ATTN_SCALE 0.10206207262f   // 1/sqrt(96)

typedef _Float16 vh8  __attribute__((ext_vector_type(8)));
typedef _Float16 vh16 __attribute__((ext_vector_type(16)));
typedef float    vf8  __attribute__((ext_vector_type(8)));

// ---------------- WMMA tile helpers (CDNA5 16x16x32 f16, wave32) ----------------
// A (16x32, row-major src, row = lane&15): halves {kb..kb+7} and {kb+16..kb+23},
// kb = (lane<16) ? 0 : 8   (ISA 7.12.2, 16-bit A-matrix table)
__device__ __forceinline__ vh16 load_a_16x32(const _Float16* base, int ld, int lane) {
  int r  = lane & 15;
  int kb = (lane < 16) ? 0 : 8;
  const _Float16* p = base + (size_t)r * ld + kb;
  vh8 lo = *(const vh8*)(p);
  vh8 hi = *(const vh8*)(p + 16);
  return __builtin_shufflevector(lo, hi, 0,1,2,3,4,5,6,7,8,9,10,11,12,13,14,15);
}
// B (32x16) from Bt stored [N][K] row-major: lane n = lane&15, halves {kb..kb+7} and
// {kb+16..kb+23}, kb = (lane<16) ? 0 : 8   (same k order as A)
__device__ __forceinline__ vh16 load_b_32x16(const _Float16* base, int ld, int lane) {
  int n  = lane & 15;
  int kb = (lane < 16) ? 0 : 8;
  const _Float16* p = base + (size_t)n * ld + kb;
  vh8 lo = *(const vh8*)(p);
  vh8 hi = *(const vh8*)(p + 16);
  return __builtin_shufflevector(lo, hi, 0,1,2,3,4,5,6,7,8,9,10,11,12,13,14,15);
}
__device__ __forceinline__ vf8 wmma_f16(vh16 a, vh16 b, vf8 c) {
  return __builtin_amdgcn_wmma_f32_16x16x32_f16(false, a, false, b, (short)0, c, false, false);
}
__device__ __forceinline__ float gelu_exact(float x) {
  return 0.5f * x * (1.0f + erff(x * 0.70710678118654752f));
}

// ---------------- LayerNorm (f32 in -> f16 out) ----------------
__global__ void __launch_bounds__(256)
ln_to_f16(const float* __restrict__ x, const float* __restrict__ g,
          const float* __restrict__ b, _Float16* __restrict__ out, int Dn) {
  int row = blockIdx.x, tid = threadIdx.x;
  const float* xr = x + (size_t)row * Dn;
  __shared__ float red[256];
  float s = 0.f;
  for (int i = tid; i < Dn; i += 256) s += xr[i];
  red[tid] = s; __syncthreads();
  for (int off = 128; off; off >>= 1) { if (tid < off) red[tid] += red[tid + off]; __syncthreads(); }
  float mean = red[0] / Dn;
  __syncthreads();
  float v = 0.f;
  for (int i = tid; i < Dn; i += 256) { float d = xr[i] - mean; v += d * d; }
  red[tid] = v; __syncthreads();
  for (int off = 128; off; off >>= 1) { if (tid < off) red[tid] += red[tid + off]; __syncthreads(); }
  float rstd = rsqrtf(red[0] / Dn + 1e-6f);
  _Float16* orow = out + (size_t)row * Dn;
  for (int i = tid; i < Dn; i += 256)
    orow[i] = (_Float16)((xr[i] - mean) * rstd * g[i] + b[i]);
}

// ---------------- f32 [R][C] -> f16 transposed [C][R] (batched via blockIdx.y) ----
__global__ void __launch_bounds__(256)
convT(const float* __restrict__ src, _Float16* __restrict__ dst, int R, int C) {
  size_t bo = (size_t)blockIdx.y * R * C;
  src += bo; dst += bo;
  int n = R * C;
  for (int i = blockIdx.x * 256 + threadIdx.x; i < n; i += gridDim.x * 256) {
    int r = i / C, c = i % C;
    dst[(size_t)c * R + r] = (_Float16)src[i];
  }
}

// ---------------- Generic WMMA GEMM: C = A(f16,MxK) @ Bt(f16,[N][K])^T ------------
// One wave -> 16 rows x (NT*16) cols tile (A-tile reused NT times per K-step).
// Optional bias, residual, f32 out, f16 out, transposed f16 out (V heads), GELU.
// Batched via blockIdx.y with strides.
template<int NT>
__global__ void __launch_bounds__(256)
gemm_f16(const _Float16* __restrict__ A, int lda, long aB,
         const _Float16* __restrict__ Bt, int ldb, long bB,
         const float* __restrict__ bias, long biasB,
         float* __restrict__ Cf, int ldc, long cB,
         const float* __restrict__ Res, int ldr, long rB,
         _Float16* __restrict__ Oh, int ldh, long hB,
         _Float16* __restrict__ Ot, int ldt, long tB, int tStart,
         int K, int tilesNg, int totalTiles, int act) {
  int wave = threadIdx.x >> 5, lane = threadIdx.x & 31;
  int id = blockIdx.x * 8 + wave;
  if (id >= totalTiles) return;
  int z  = blockIdx.y;
  int mt = id / tilesNg, ng = id % tilesNg;

  const _Float16* Az = A + (size_t)z * aB;
  const _Float16* Bz = Bt + (size_t)z * bB;
  const _Float16* Ar = Az + (size_t)mt * 16 * lda;
  const _Float16* Bg = Bz + (size_t)(ng * NT * 16) * ldb;

  vf8 cc[NT];
  #pragma unroll
  for (int t = 0; t < NT; ++t) cc[t] = (vf8){};

  for (int k = 0; k < K; k += 32) {
    vh16 a = load_a_16x32(Ar + k, lda, lane);
    #pragma unroll
    for (int t = 0; t < NT; ++t)
      cc[t] = wmma_f16(a, load_b_32x16(Bg + (size_t)t * 16 * ldb + k, ldb, lane), cc[t]);
  }

  const float* biasZ = bias ? bias + (size_t)z * biasB : nullptr;
  float*       CfZ   = Cf   ? Cf   + (size_t)z * cB    : nullptr;
  const float* ResZ  = Res  ? Res  + (size_t)z * rB    : nullptr;
  _Float16*    OhZ   = Oh   ? Oh   + (size_t)z * hB    : nullptr;
  _Float16*    OtZ   = (Ot && z >= tStart) ? Ot + (size_t)(z - tStart) * tB : nullptr;

  int rbase = mt * 16 + ((lane >> 4) << 3);
  #pragma unroll
  for (int t = 0; t < NT; ++t) {
    int col = (ng * NT + t) * 16 + (lane & 15);
    float bv = biasZ ? biasZ[col] : 0.f;
    #pragma unroll
    for (int v = 0; v < 8; ++v) {
      float val = cc[t][v] + bv;
      if (act == 1) val = gelu_exact(val);
      int row = rbase + v;
      if (ResZ) val += ResZ[(size_t)row * ldr + col];
      if (CfZ)  CfZ[(size_t)row * ldc + col] = val;
      if (OhZ)  OhZ[(size_t)row * ldh + col] = (_Float16)val;
      if (OtZ)  OtZ[(size_t)col * ldt + row] = (_Float16)val;
    }
  }
}

// ---------------- Attention: 1 wave per (head, 16 query rows) --------------------
// Pass1: online row max/sum over all keys (scores via WMMA, recomputed)
// Pass2: normalized probs -> global (mandatory output) + f16 via LDS -> probs@V WMMA
__global__ void __launch_bounds__(256)
attn_kernel(const _Float16* __restrict__ qkvh, const _Float16* __restrict__ vt,
            float* __restrict__ attnw, _Float16* __restrict__ attn_m) {
  __shared__ _Float16 pls[8][16 * 32];
  int wave = threadIdx.x >> 5, lane = threadIdx.x & 31;
  int rowblk = blockIdx.x * 8 + wave;      // 0..127
  int head   = blockIdx.y;                 // 0..7  (g*2+q)
  int g = head >> 1, qi = head & 1;
  int row0 = rowblk * 16;

  const _Float16* qb = qkvh + ((size_t)head * PSEQ + row0) * HDIM;
  const _Float16* kb = qkvh + (size_t)(NQH + g) * PSEQ * HDIM;
  const _Float16* vb = vt   + (size_t)g * HDIM * PSEQ;

  vh16 a0 = load_a_16x32(qb +  0, HDIM, lane);
  vh16 a1 = load_a_16x32(qb + 32, HDIM, lane);
  vh16 a2 = load_a_16x32(qb + 64, HDIM, lane);

  float mrow[8], srow[8];
  #pragma unroll
  for (int v = 0; v < 8; ++v) { mrow[v] = -1e30f; srow[v] = 0.f; }

  // ---- pass 1: row max + sum of exp ----
  for (int jt = 0; jt < PSEQ / 16; ++jt) {
    const _Float16* kt = kb + (size_t)jt * 16 * HDIM;
    vf8 c = {};
    c = wmma_f16(a0, load_b_32x16(kt +  0, HDIM, lane), c);
    c = wmma_f16(a1, load_b_32x16(kt + 32, HDIM, lane), c);
    c = wmma_f16(a2, load_b_32x16(kt + 64, HDIM, lane), c);
    #pragma unroll
    for (int v = 0; v < 8; ++v) {
      float val = c[v] * ATTN_SCALE;
      float nm  = fmaxf(mrow[v], val);
      srow[v] = srow[v] * __expf(mrow[v] - nm) + __expf(val - nm);
      mrow[v] = nm;
    }
  }
  // reduce (m,s) across the 16 lanes of each half-wave
  #pragma unroll
  for (int mask = 1; mask < 16; mask <<= 1) {
    #pragma unroll
    for (int v = 0; v < 8; ++v) {
      float mo = __shfl_xor(mrow[v], mask, 32);
      float so = __shfl_xor(srow[v], mask, 32);
      float nm = fmaxf(mrow[v], mo);
      srow[v] = srow[v] * __expf(mrow[v] - nm) + so * __expf(mo - nm);
      mrow[v] = nm;
    }
  }
  float rinv[8];
  #pragma unroll
  for (int v = 0; v < 8; ++v) rinv[v] = 1.f / srow[v];

  // ---- pass 2: write probs, accumulate probs @ V ----
  vf8 acc[6];
  #pragma unroll
  for (int nt = 0; nt < 6; ++nt) acc[nt] = (vf8){};
  float* aw = attnw + (size_t)qi * GKV * PSEQ * PSEQ + (size_t)(g * PSEQ + row0) * PSEQ;
  _Float16* pw = &pls[wave][0];

  for (int kb2 = 0; kb2 < PSEQ / 32; ++kb2) {
    #pragma unroll
    for (int sub = 0; sub < 2; ++sub) {
      int jt = kb2 * 2 + sub;
      const _Float16* kt = kb + (size_t)jt * 16 * HDIM;
      vf8 c = {};
      c = wmma_f16(a0, load_b_32x16(kt +  0, HDIM, lane), c);
      c = wmma_f16(a1, load_b_32x16(kt + 32, HDIM, lane), c);
      c = wmma_f16(a2, load_b_32x16(kt + 64, HDIM, lane), c);
      int colg = jt * 16 + (lane & 15);
      #pragma unroll
      for (int v = 0; v < 8; ++v) {
        int rv = v + ((lane >> 4) << 3);
        float pv = __expf(c[v] * ATTN_SCALE - mrow[v]) * rinv[v];
        aw[(size_t)rv * PSEQ + colg] = pv;                       // softmax output
        pw[rv * 32 + sub * 16 + (lane & 15)] = (_Float16)pv;     // C-layout -> A-layout via LDS
      }
    }
    vh16 pa = load_a_16x32(pw, 32, lane);
    const _Float16* vtk = vb + (size_t)kb2 * 32;
    #pragma unroll
    for (int nt = 0; nt < 6; ++nt)
      acc[nt] = wmma_f16(pa, load_b_32x16(vtk + (size_t)nt * 16 * PSEQ, PSEQ, lane), acc[nt]);
  }

  // store attn output pre-merged as [g][p][q*H + h] f16
  _Float16* ob = attn_m + ((size_t)g * PSEQ + row0) * (QPG * HDIM) + qi * HDIM;
  #pragma unroll
  for (int nt = 0; nt < 6; ++nt)
    #pragma unroll
    for (int v = 0; v < 8; ++v) {
      int rv = v + ((lane >> 4) << 3);
      ob[(size_t)rv * (QPG * HDIM) + nt * 16 + (lane & 15)] = (_Float16)acc[nt][v];
    }
}

// ======================= host launcher =======================
extern "C" void kernel_launch(void* const* d_in, const int* in_sizes, int n_in,
                              void* d_out, int out_size, void* d_ws, size_t ws_size,
                              hipStream_t stream) {
  (void)in_sizes; (void)n_in; (void)out_size; (void)ws_size;
  const float* x     = (const float*)d_in[0];
  const float* ln1_g = (const float*)d_in[1];
  const float* ln1_b = (const float*)d_in[2];
  const float* Wq    = (const float*)d_in[3];
  const float* bq    = (const float*)d_in[4];
  const float* Wk    = (const float*)d_in[5];
  const float* bk    = (const float*)d_in[6];
  const float* Wv    = (const float*)d_in[7];
  const float* bv    = (const float*)d_in[8];
  const float* Wm    = (const float*)d_in[9];
  const float* bm    = (const float*)d_in[10];
  const float* Wp    = (const float*)d_in[11];
  const float* bp    = (const float*)d_in[12];
  const float* ln2_g = (const float*)d_in[13];
  const float* ln2_b = (const float*)d_in[14];
  const float* W1    = (const float*)d_in[15];
  const float* b1    = (const float*)d_in[16];
  const float* W2    = (const float*)d_in[17];
  const float* b2    = (const float*)d_in[18];

  float* outp  = (float*)d_out;                       // [P, D]
  float* attnw = outp + (size_t)PSEQ * DEMB;          // [Q, G*P, P]

  char* w = (char*)d_ws;
  auto carve = [&](size_t bytes) -> char* {
    char* p = w; w += (bytes + 255) & ~(size_t)255; return p;
  };
  _Float16* xh     = (_Float16*)carve((size_t)PSEQ * DEMB * 2);
  _Float16* yh     = (_Float16*)carve((size_t)PSEQ * DEMB * 2);
  _Float16* wqkvt  = (_Float16*)carve((size_t)16 * HDIM * DEMB * 2);
  float*    bqkv   = (float*)   carve((size_t)16 * HDIM * 4);
  _Float16* qkvh   = (_Float16*)carve((size_t)16 * PSEQ * HDIM * 2);
  _Float16* vt     = (_Float16*)carve((size_t)GKV * HDIM * PSEQ * 2);
  _Float16* wmt    = (_Float16*)carve((size_t)GKV * HDIM * (QPG * HDIM) * 2);
  _Float16* attn_m = (_Float16*)carve((size_t)GKV * PSEQ * (QPG * HDIM) * 2);
  _Float16* wpt    = (_Float16*)carve((size_t)DEMB * (GKV * HDIM) * 2);
  _Float16* conc   = (_Float16*)carve((size_t)PSEQ * (GKV * HDIM) * 2);
  float*    out1   = (float*)   carve((size_t)PSEQ * DEMB * 4);
  _Float16* w1t    = (_Float16*)carve((size_t)HIDD * DEMB * 2);
  _Float16* hh     = (_Float16*)carve((size_t)PSEQ * HIDD * 2);
  _Float16* w2t    = (_Float16*)carve((size_t)DEMB * HIDD * 2);

  // LN1: x -> xh (f16)
  ln_to_f16<<<PSEQ, 256, 0, stream>>>(x, ln1_g, ln1_b, xh, DEMB);

  // weight convert+transpose to f16 [N][K]
  convT<<<dim3(288, 8),  256, 0, stream>>>(Wq, wqkvt,                    DEMB, HDIM);
  convT<<<dim3(288, 4),  256, 0, stream>>>(Wk, wqkvt +  8 * HDIM * DEMB, DEMB, HDIM);
  convT<<<dim3(288, 4),  256, 0, stream>>>(Wv, wqkvt + 12 * HDIM * DEMB, DEMB, HDIM);
  convT<<<dim3(72, 4),   256, 0, stream>>>(Wm, wmt, QPG * HDIM, HDIM);
  convT<<<dim3(1152, 1), 256, 0, stream>>>(Wp, wpt, GKV * HDIM, DEMB);
  convT<<<dim3(9216, 1), 256, 0, stream>>>(W1, w1t, DEMB, HIDD);
  convT<<<dim3(9216, 1), 256, 0, stream>>>(W2, w2t, HIDD, DEMB);

  // pack qkv biases [16*96]
  hipMemcpyAsync(bqkv,        bq, (size_t)NQH * HDIM * 4, hipMemcpyDeviceToDevice, stream);
  hipMemcpyAsync(bqkv +  768, bk, (size_t)GKV * HDIM * 4, hipMemcpyDeviceToDevice, stream);
  hipMemcpyAsync(bqkv + 1152, bv, (size_t)GKV * HDIM * 4, hipMemcpyDeviceToDevice, stream);

  // QKV projections: 16 heads batched; V heads (z>=12) also write vt transposed
  gemm_f16<2><<<dim3(48, 16), 256, 0, stream>>>(
      xh, DEMB, 0L, wqkvt, DEMB, (long)HDIM * DEMB, bqkv, (long)HDIM,
      nullptr, 0, 0L, nullptr, 0, 0L,
      qkvh, HDIM, (long)PSEQ * HDIM,
      vt, PSEQ, (long)HDIM * PSEQ, 12,
      DEMB, 3, 384, 0);

  // attention (writes softmax probs to d_out and pre-merged attn to attn_m)
  attn_kernel<<<dim3(16, 8), 256, 0, stream>>>(qkvh, vt, attnw, attn_m);

  // merge: per-g [P,192] @ [192,96] -> conc[p][g*96+h]
  gemm_f16<2><<<dim3(48, 4), 256, 0, stream>>>(
      attn_m, QPG * HDIM, (long)PSEQ * QPG * HDIM, wmt, QPG * HDIM, (long)HDIM * QPG * HDIM,
      bm, (long)HDIM,
      nullptr, 0, 0L, nullptr, 0, 0L,
      conc, GKV * HDIM, (long)HDIM,
      nullptr, 0, 0L, 0,
      QPG * HDIM, 3, 384, 0);

  // out1 = x + conc @ Wp + bp   (N=768 -> 12 groups of 64)
  gemm_f16<4><<<dim3(192, 1), 256, 0, stream>>>(
      conc, GKV * HDIM, 0L, wpt, GKV * HDIM, 0L, bp, 0L,
      out1, DEMB, 0L, x, DEMB, 0L,
      nullptr, 0, 0L, nullptr, 0, 0L, 0,
      GKV * HDIM, 12, 1536, 0);

  // LN2 -> yh
  ln_to_f16<<<PSEQ, 256, 0, stream>>>(out1, ln2_g, ln2_b, yh, DEMB);

  // MLP1 with exact GELU -> hh (f16)   (N=3072 -> 48 groups of 64)
  gemm_f16<4><<<dim3(768, 1), 256, 0, stream>>>(
      yh, DEMB, 0L, w1t, DEMB, 0L, b1, 0L,
      nullptr, 0, 0L, nullptr, 0, 0L,
      hh, HIDD, 0L,
      nullptr, 0, 0L, 0,
      DEMB, 48, 6144, 1);

  // MLP2 + residual out1 -> final out (d_out)   (N=768 -> 12 groups of 64)
  gemm_f16<4><<<dim3(192, 1), 256, 0, stream>>>(
      hh, HIDD, 0L, w2t, HIDD, 0L, b2, 0L,
      outp, DEMB, 0L, out1, DEMB, 0L,
      nullptr, 0, 0L, nullptr, 0, 0L, 0,
      HIDD, 12, 1536, 0);
}
